// V2V_UNet_46016279609733
// MI455X (gfx1250) — matrix operand order repaired and checked by reading, not compiled, not run on hardware
//
#include <hip/hip_runtime.h>
#include <math.h>

typedef __attribute__((ext_vector_type(16))) _Float16 v16h;
typedef __attribute__((ext_vector_type(8)))  float    v8f;

#define NAGENT 5
#define EPSBN 1e-5f

static __host__ __device__ inline int c32(int c) { return (c + 31) & ~31; }

// ---------------------------------------------------------------- layout conversion
// x fp32 NCHW (13 ch) -> f16 NHWC padded to 32 ch (zeros)
__global__ void k_x_to_nhwc(const float* __restrict__ x, _Float16* __restrict__ o,
                            int Cin, int Cpad, int HW, size_t total) {
  size_t i = (size_t)blockIdx.x * blockDim.x + threadIdx.x;
  if (i >= total) return;
  int c = (int)(i % (size_t)Cpad);
  size_t r = i / (size_t)Cpad;
  int p = (int)(r % (size_t)HW);
  int n = (int)(r / (size_t)HW);
  o[i] = (c < Cin) ? (_Float16)x[((size_t)n * Cin + c) * HW + p] : (_Float16)0.f;
}

// weights (O,I,3,3) fp32 -> f16 [Cout][tap 0..8][ci 0..Cpad) (tap-major K, zero-padded ci)
__global__ void k_pack_w(const float* __restrict__ w, _Float16* __restrict__ o,
                         int Cin, int Cpad, size_t total) {
  size_t i = (size_t)blockIdx.x * blockDim.x + threadIdx.x;
  if (i >= total) return;
  int ci = (int)(i % (size_t)Cpad);
  size_t r = i / (size_t)Cpad;
  int tap = (int)(r % 9);
  int co  = (int)(r / 9);
  o[i] = (ci < Cin) ? (_Float16)w[((size_t)co * Cin + ci) * 9 + tap] : (_Float16)0.f;
}

// ------------------------------------------------ implicit-GEMM 3x3 conv (NHWC), WMMA f16
// GEMM: M = Cout (A = weights), N = pixels (B = im2col in LDS), K = 9*Cpad (tap-major).
// Block = 256 thr = 8 waves; block tile = 64 px x 64 Cout; wave tile = 16 px x 32 Cout.
// LDS double-buffered: one barrier per 32-K step; next step's activation chunk is loaded
// global->VGPR while the current step's 2 WMMAs execute, then ds_store'd to the spare buffer.
// K ordering is tap-major and contiguous, so the weight offset of step s is simply s*32.
__global__ __launch_bounds__(256)
void k_conv3x3_wmma(const _Float16* __restrict__ in,   // [Nimg][H][W][Cpad] f16
                    const _Float16* __restrict__ wts,  // [Cout][9*Cpad] f16
                    float* __restrict__ out,           // [Nimg][H][W][Cout] fp32
                    int Cpad, int Cout, int H, int W, int lw) {
  const int HW   = H * W;
  const int Kpad = 9 * Cpad;
  const int n    = blockIdx.y;
  const int px0  = blockIdx.x * 64;
  const int co0  = blockIdx.z * 64;
  const int t    = threadIdx.x;
  const int lane = t & 31;
  const int wave = t >> 5;
  const int pxt  = wave & 3;   // pixel sub-tile 0..3
  const int cot  = wave >> 2;  // cout sub-tile 0..1 (32 couts each)

  __shared__ alignas(64) _Float16 sB[2][64 * 32];

  v8f acc0, acc1;
#pragma unroll
  for (int i = 0; i < 8; ++i) { acc0[i] = 0.f; acc1[i] = 0.f; }

  const int laneM = lane & 15;
  const int hi    = lane >> 4;
  // A layout (16-bit 16x32): lanes<16 hold K 0-7 & 16-23; lanes>=16 hold K 8-15 & 24-31
  const _Float16* wrow0 = wts + (size_t)(co0 + cot * 32 + laneM) * Kpad + hi * 8;
  const _Float16* wrow1 = wrow0 + (size_t)16 * Kpad;
  const _Float16* inn   = in + (size_t)n * HW * Cpad;

  // fill mapping: 4 threads per pixel, 8 contiguous channels each
  const int pxf = t >> 2;
  const int cio = (t & 3) * 8;
  const int ph  = (px0 + pxf) >> lw;
  const int pw  = (px0 + pxf) & (W - 1);
  const int sAddr = pxf * 32 + cio;      // this thread's LDS slot (halves)
  const int bAddr = (pxt * 16 + laneM) * 32 + hi * 8;  // this lane's B strip (halves)

  // ---- fill-state (next K-step to fetch): tap + channel-block, border validity
  int f_tap = 0, f_cb = 0;
  int fh = ph - 1, fw = pw - 1;  // tap 0 -> (dh,dw)=(-1,-1)
  bool f_ok = ((unsigned)fh < (unsigned)H) && ((unsigned)fw < (unsigned)W);
  const _Float16* f_gp = inn + (size_t)(fh * W + fw) * Cpad + cio;

  // prologue: fill step 0 into buffer 0
  {
    uint4 v = make_uint4(0u, 0u, 0u, 0u);
    if (f_ok) v = *(const uint4*)(f_gp + f_cb);
    *(uint4*)&sB[0][sAddr] = v;
    f_cb += 32;
    if (f_cb == Cpad) {
      f_cb = 0; ++f_tap;
      int q3 = f_tap / 3;
      int h2 = ph + q3 - 1, w2 = pw + (f_tap - q3 * 3) - 1;
      f_ok = ((unsigned)h2 < (unsigned)H) && ((unsigned)w2 < (unsigned)W);
      f_gp = inn + (size_t)(h2 * W + w2) * Cpad + cio;
    }
  }
  __syncthreads();

  const int S = Kpad >> 5;  // total 32-K steps
  for (int s = 0; s < S; ++s) {
    const int  buf  = s & 1;
    const bool more = (s + 1 < S);

    // start next step's global fetch early (overlaps the WMMAs below)
    uint4 nv = make_uint4(0u, 0u, 0u, 0u);
    if (more && f_ok) nv = *(const uint4*)(f_gp + f_cb);

    union Frag { v16h v; uint4 q[2]; } A0, A1, Bm;
    const _Float16* wp0 = wrow0 + (size_t)s * 32;
    const _Float16* wp1 = wrow1 + (size_t)s * 32;
    __builtin_prefetch((const void*)(wp0 + 64), 0, 0);  // weights two chunks (K-steps) ahead
    A0.q[0] = *(const uint4*)(wp0);
    A0.q[1] = *(const uint4*)(wp0 + 16);
    A1.q[0] = *(const uint4*)(wp1);
    A1.q[1] = *(const uint4*)(wp1 + 16);
    // B layout (32x16): lanes<16 -> K 0-7 & 16-23, lanes>=16 -> K 8-15 & 24-31, col = lane&15
    const _Float16* bp = &sB[buf][bAddr];
    Bm.q[0] = *(const uint4*)(bp);
    Bm.q[1] = *(const uint4*)(bp + 16);

    acc0 = __builtin_amdgcn_wmma_f32_16x16x32_f16(false, A0.v, false, Bm.v,
                                                  (short)0, acc0, false, false);
    acc1 = __builtin_amdgcn_wmma_f32_16x16x32_f16(false, A1.v, false, Bm.v,
                                                  (short)0, acc1, false, false);

    if (more) {
      *(uint4*)&sB[buf ^ 1][sAddr] = nv;
      f_cb += 32;
      if (f_cb == Cpad) {
        f_cb = 0; ++f_tap;
        if (f_tap < 9) {
          int q3 = f_tap / 3;
          int h2 = ph + q3 - 1, w2 = pw + (f_tap - q3 * 3) - 1;
          f_ok = ((unsigned)h2 < (unsigned)H) && ((unsigned)w2 < (unsigned)W);
          f_gp = inn + (size_t)(h2 * W + w2) * Cpad + cio;
        }
      }
    }
    __syncthreads();  // single barrier per K-step
  }

  // D layout: VGPR r, lane l -> M = r + (l>=16 ? 8:0), N = l&15. NHWC: 8 contiguous couts/lane.
  const int px  = px0 + pxt * 16 + laneM;
  const int coA = co0 + cot * 32 + hi * 8;
  float* op = out + ((size_t)n * HW + px) * Cout + coA;
  union { v8f v; float4 f[2]; } s0, s1;
  s0.v = acc0; s1.v = acc1;
  *(float4*)(op)      = s0.f[0];
  *(float4*)(op + 4)  = s0.f[1];
  *(float4*)(op + 16) = s1.f[0];
  *(float4*)(op + 20) = s1.f[1];
}

// ---------------------------------------------------------------- batchnorm (deterministic 2-stage)
__global__ __launch_bounds__(256)
void k_bn_partial(const float* __restrict__ raw, float* __restrict__ part,
                  int C, size_t Ptot) {
  const int t = threadIdx.x;
  const size_t p0 = (size_t)blockIdx.x * 2048;
  const size_t pend = (p0 + 2048 < Ptot) ? p0 + 2048 : Ptot;
  __shared__ float ls[256], lss[256];
  const int G    = (C < 256) ? (256 / C) : 1;
  const int c    = (C < 256) ? (t % C) : t;
  const int psub = (C < 256) ? (t / C) : 0;
  for (int cc = c; cc < C; cc += 256) {
    float s = 0.f, ss = 0.f;
    for (size_t p = p0 + psub; p < pend; p += G) {
      float v = raw[p * C + cc];
      s += v; ss += v * v;
    }
    ls[t] = s; lss[t] = ss;
    __syncthreads();
    if (psub == 0) {
      for (int g = 1; g < G; ++g) { s += ls[t + g * C]; ss += lss[t + g * C]; }
      part[(size_t)blockIdx.x * 2 * C + cc]     = s;
      part[(size_t)blockIdx.x * 2 * C + C + cc] = ss;
    }
    __syncthreads();
  }
}

__global__ void k_bn_finalize(const float* __restrict__ part, const float* __restrict__ g,
                              const float* __restrict__ be, float* __restrict__ scale,
                              float* __restrict__ shift, int C, int nblk, float cnt) {
  int c = blockIdx.x * blockDim.x + threadIdx.x;
  if (c >= C) return;
  float s = 0.f, ss = 0.f;
  for (int b = 0; b < nblk; ++b) {
    s  += part[(size_t)b * 2 * C + c];
    ss += part[(size_t)b * 2 * C + C + c];
  }
  float m = s / cnt;
  float v = ss / cnt - m * m;
  float a = g[c] * rsqrtf(v + EPSBN);
  scale[c] = a;
  shift[c] = be[c] - m * a;
}

__global__ void k_bn_relu(const float* __restrict__ raw, const float* __restrict__ scale,
                          const float* __restrict__ shift, _Float16* __restrict__ out,
                          int C, size_t total) {
  size_t i = (size_t)blockIdx.x * blockDim.x + threadIdx.x;
  if (i >= total) return;
  int c = (int)(i % (size_t)C);
  float y = raw[i] * scale[c] + shift[c];
  out[i] = (_Float16)fmaxf(y, 0.f);
}

// ---------------------------------------------------------------- pool / upsample+concat (NHWC)
__global__ void k_maxpool(const _Float16* __restrict__ in, _Float16* __restrict__ out,
                          int C, int H, int W, size_t total) {
  size_t i = (size_t)blockIdx.x * blockDim.x + threadIdx.x;
  if (i >= total) return;
  int Wo = W >> 1;
  int c  = (int)(i % (size_t)C); size_t r = i / (size_t)C;
  int ow = (int)(r % (size_t)Wo); r /= (size_t)Wo;
  int oh = (int)(r % (size_t)(H >> 1));
  int n  = (int)(r / (size_t)(H >> 1));
  const _Float16* p = in + (((size_t)n * H + 2 * oh) * W + 2 * ow) * C + c;
  float a = (float)p[0], b = (float)p[C];
  float d = (float)p[(size_t)W * C], e = (float)p[(size_t)W * C + C];
  out[i] = (_Float16)fmaxf(fmaxf(a, b), fmaxf(d, e));
}

__global__ void k_upcat(const _Float16* __restrict__ x2, const _Float16* __restrict__ x1,
                        _Float16* __restrict__ cat, int C2, int C1, int Ho, int Wo, size_t total) {
  size_t i = (size_t)blockIdx.x * blockDim.x + threadIdx.x;
  if (i >= total) return;
  const int Ct = C1 + C2;
  int c  = (int)(i % (size_t)Ct); size_t r = i / (size_t)Ct;
  int ow = (int)(r % (size_t)Wo); r /= (size_t)Wo;
  int oh = (int)(r % (size_t)Ho);
  int n  = (int)(r / (size_t)Ho);
  if (c < C2) {
    cat[i] = x2[(((size_t)n * Ho + oh) * Wo + ow) * C2 + c];
  } else {
    int Hi = Ho >> 1, Wi = Wo >> 1;
    float ch = (float)oh * (float)(Hi - 1) / (float)(Ho - 1);
    float cw = (float)ow * (float)(Wi - 1) / (float)(Wo - 1);
    int i0 = (int)floorf(ch); int i1 = min(i0 + 1, Hi - 1); float hf = ch - (float)i0;
    int j0 = (int)floorf(cw); int j1 = min(j0 + 1, Wi - 1); float wf = cw - (float)j0;
    const _Float16* p = x1 + (size_t)n * Hi * Wi * C1 + (c - C2);
    float v00 = (float)p[((size_t)i0 * Wi + j0) * C1];
    float v01 = (float)p[((size_t)i0 * Wi + j1) * C1];
    float v10 = (float)p[((size_t)i1 * Wi + j0) * C1];
    float v11 = (float)p[((size_t)i1 * Wi + j1) * C1];
    float v = (1.f - hf) * ((1.f - wf) * v00 + wf * v01) + hf * ((1.f - wf) * v10 + wf * v11);
    cat[i] = (_Float16)v;
  }
}

// ---------------------------------------------------------------- V2V fusion (NHWC)
template <typename T>
static __device__ __forceinline__ float bil_tap(const T* img, int H, int W, int Cs,
                                                float x, float y) {
  int x0 = (int)floorf(x), y0 = (int)floorf(y);
  float wx = x - (float)x0, wy = y - (float)y0;
  float acc = 0.f;
#pragma unroll
  for (int dy = 0; dy < 2; ++dy)
#pragma unroll
    for (int dx = 0; dx < 2; ++dx) {
      int xi = x0 + dx, yi = y0 + dy;
      float wgt = (dx ? wx : 1.f - wx) * (dy ? wy : 1.f - wy);
      if (xi >= 0 && xi < W && yi >= 0 && yi < H)
        acc += wgt * (float)img[(size_t)(yi * W + xi) * Cs];
    }
  return acc;
}

__global__ void k_warp_rot(const _Float16* __restrict__ x4, const float* __restrict__ tm,
                           float* __restrict__ tmp, int B, int C, int H, int W, size_t total) {
  size_t idx = (size_t)blockIdx.x * blockDim.x + threadIdx.x;
  if (idx >= total) return;
  int c = (int)(idx % (size_t)C); size_t r = idx / (size_t)C;
  int w = (int)(r % (size_t)W); r /= (size_t)W;
  int h = (int)(r % (size_t)H); r /= (size_t)H;
  int j = (int)(r % NAGENT); r /= NAGENT;
  int i = (int)(r % NAGENT);
  int b = (int)(r / NAGENT);
  const float* T = tm + (((size_t)b * NAGENT + i) * NAGENT + j) * 16;
  float X = (2.f * w + 1.f) / (float)W - 1.f;
  float Y = (2.f * h + 1.f) / (float)H - 1.f;
  float gx = T[0] * X + T[1] * Y;
  float gy = T[4] * X + T[5] * Y;
  float x = (gx + 1.f) * W * 0.5f - 0.5f;
  float y = (gy + 1.f) * H * 0.5f - 0.5f;
  const _Float16* img = x4 + (size_t)(j * B + b) * H * W * C + c;
  tmp[idx] = bil_tap(img, H, W, C, x, y);
}

__global__ void k_fuse_cat(const _Float16* __restrict__ x4, const float* __restrict__ tmp,
                           const float* __restrict__ tm, _Float16* __restrict__ cat,
                           int B, int C, int H, int W, size_t total) {
  size_t idx = (size_t)blockIdx.x * blockDim.x + threadIdx.x;
  if (idx >= total) return;
  int c = (int)(idx % (size_t)C); size_t r = idx / (size_t)C;
  int w = (int)(r % (size_t)W); r /= (size_t)W;
  int h = (int)(r % (size_t)H); r /= (size_t)H;
  int i = (int)(r % NAGENT);
  int b = (int)(r / NAGENT);
  size_t HW = (size_t)H * W;
  size_t p  = (size_t)h * W + w;
  float self = (float)x4[((size_t)(i * B + b) * HW + p) * C + c];
  float acc = self;
  for (int j = 0; j < NAGENT; ++j) {
    if (j == i) continue;
    const float* T = tm + (((size_t)b * NAGENT + i) * NAGENT + j) * 16;
    float xt =  4.f * T[3] / 128.f;
    float yt = -4.f * T[7] / 128.f;
    float x = (float)w + xt * W * 0.5f;
    float y = (float)h + yt * H * 0.5f;
    const float* img = tmp + ((((size_t)b * NAGENT + i) * NAGENT + j) * HW) * C + c;
    acc += bil_tap(img, H, W, C, x, y);
  }
  acc *= (1.f / (float)NAGENT);
  size_t nimg = (size_t)b * NAGENT + i;
  _Float16* base = cat + (nimg * HW + p) * (size_t)(2 * C);
  base[c]     = (_Float16)self;
  base[C + c] = (_Float16)acc;
}

__global__ void k_gru_gate(const float* __restrict__ gi, const float* __restrict__ bih,
                           const float* __restrict__ bhh, _Float16* __restrict__ feat,
                           int B, int C, int HW, size_t total) {
  size_t idx = (size_t)blockIdx.x * blockDim.x + threadIdx.x;
  if (idx >= total) return;
  int c = (int)(idx % (size_t)C); size_t r = idx / (size_t)C;
  int p = (int)(r % (size_t)HW);
  int nimg = (int)(r / (size_t)HW);
  const float* gp = gi + ((size_t)nimg * HW + p) * (size_t)(3 * C);
  float rr = gp[c] + bih[c] + bhh[c];
  rr = 1.f / (1.f + expf(-rr));
  float zz = gp[C + c] + bih[C + c] + bhh[C + c];
  zz = 1.f / (1.f + expf(-zz));
  float nn = tanhf(gp[2 * C + c] + bih[2 * C + c] + rr * bhh[2 * C + c]);
  float hv = (1.f - zz) * nn;
  int b = nimg / NAGENT, i = nimg - b * NAGENT;
  feat[((size_t)(i * B + b) * HW + p) * C + c] = (_Float16)hv;
}

__global__ void k_outc(const _Float16* __restrict__ in, const float* __restrict__ w,
                       const float* __restrict__ b, float* __restrict__ out,
                       int HW, size_t total) {
  size_t idx = (size_t)blockIdx.x * blockDim.x + threadIdx.x;
  if (idx >= total) return;
  int p = (int)(idx % (size_t)HW);
  int n = (int)(idx / (size_t)HW);
  const _Float16* ip = in + ((size_t)n * HW + p) * 64;
  float v[64];
#pragma unroll
  for (int c = 0; c < 64; ++c) v[c] = (float)ip[c];
  for (int cls = 0; cls < 11; ++cls) {
    float s = b[cls];
    const float* wp = w + cls * 64;
#pragma unroll
    for (int c = 0; c < 64; ++c) s += v[c] * wp[c];
    out[((size_t)n * 11 + cls) * (size_t)HW + p] = s;
  }
}

// ---------------------------------------------------------------- host driver
static inline size_t wbytes(int cin, int cout) { return (size_t)cout * 9 * c32(cin) * 2; }

extern "C" void kernel_launch(void* const* d_in, const int* in_sizes, int n_in,
                              void* d_out, int out_size, void* d_ws, size_t ws_size,
                              hipStream_t stream) {
  (void)n_in; (void)ws_size; (void)out_size;
  const int H0 = 256;
  int B = in_sizes[0] / (NAGENT * 13 * H0 * H0);
  if (B < 1) B = 1;
  const int Nimg = NAGENT * B;

  const float* x_f32 = (const float*)d_in[0];
  const float* tm    = (const float*)d_in[1];
  auto P = [&](int i) { return (const float*)d_in[3 + i]; };
  const int IDX_OUTC_W = 72, IDX_OUTC_B = 73, IDX_GRU_W = 74, IDX_GRU_BIH = 75, IDX_GRU_BHH = 76;

  struct DCfg { int cin, mid, cout, Hs; };
  const DCfg dc[9] = {
    {13, 64, 64, 256}, {64, 128, 128, 128}, {128, 256, 256, 64}, {256, 512, 512, 32},
    {512, 512, 512, 16},
    {1024, 512, 256, 32}, {512, 256, 128, 64}, {256, 128, 64, 128}, {128, 64, 64, 256}};

  char* cur = (char*)d_ws;
  auto alloc = [&](size_t bytes) -> void* {
    void* r = (void*)cur; cur += (bytes + 255) & ~(size_t)255; return r;
  };

  _Float16* wk[19];
  for (int k = 0; k < 9; ++k) {
    wk[k * 2 + 0] = (_Float16*)alloc(wbytes(dc[k].cin, dc[k].mid));
    wk[k * 2 + 1] = (_Float16*)alloc(wbytes(dc[k].mid, dc[k].cout));
  }
  wk[18] = (_Float16*)alloc(wbytes(1024, 1536));

  const size_t HW1 = (size_t)H0 * H0;
  _Float16* xin  = (_Float16*)alloc((size_t)Nimg * HW1 * 32 * 2);
  _Float16* actA = (_Float16*)alloc((size_t)Nimg * 128 * HW1 * 2);
  _Float16* actB = (_Float16*)alloc((size_t)Nimg * 128 * HW1 * 2);
  float*    raw  = (float*)   alloc((size_t)Nimg * 64  * HW1 * 4);
  _Float16* x1s  = (_Float16*)alloc((size_t)Nimg * 64  * HW1 * 2);
  _Float16* x2s  = (_Float16*)alloc((size_t)Nimg * 128 * (HW1 / 4)   * 2);
  _Float16* x3s  = (_Float16*)alloc((size_t)Nimg * 256 * (HW1 / 16)  * 2);
  _Float16* x4s  = (_Float16*)alloc((size_t)Nimg * 512 * (HW1 / 64)  * 2);
  _Float16* fms  = (_Float16*)alloc((size_t)Nimg * 512 * (HW1 / 64)  * 2);
  _Float16* x5s  = (_Float16*)alloc((size_t)Nimg * 512 * (HW1 / 256) * 2);
  float* part  = (float*)alloc((size_t)((Nimg * HW1 + 2047) / 2048) * 2 * 512 * 4);
  float* scale = (float*)alloc(2048 * 4);
  float* shift = (float*)alloc(2048 * 4);

  auto grid1 = [](size_t tot) { return dim3((unsigned)((tot + 255) / 256)); };
  auto pack = [&](const float* src, _Float16* dst, int cin, int cout) {
    size_t tot = (size_t)cout * 9 * c32(cin);
    k_pack_w<<<grid1(tot), 256, 0, stream>>>(src, dst, cin, c32(cin), tot);
  };
  auto conv = [&](const _Float16* in, const _Float16* w, float* o, int cin, int cout, int Hs) {
    dim3 g((unsigned)((Hs * Hs) / 64), (unsigned)Nimg, (unsigned)(cout / 64));
    k_conv3x3_wmma<<<g, 256, 0, stream>>>(in, w, o, c32(cin), cout, Hs, Hs, __builtin_ctz(Hs));
  };
  auto bnrelu = [&](const float* gamma, const float* beta, _Float16* o, int cch, int Hs) {
    size_t Ptot = (size_t)Nimg * Hs * Hs;
    int nblk = (int)((Ptot + 2047) / 2048);
    k_bn_partial<<<dim3((unsigned)nblk), 256, 0, stream>>>(raw, part, cch, Ptot);
    k_bn_finalize<<<grid1((size_t)cch), 256, 0, stream>>>(part, gamma, beta, scale, shift,
                                                          cch, nblk, (float)Ptot);
    size_t tot = Ptot * cch;
    k_bn_relu<<<grid1(tot), 256, 0, stream>>>(raw, scale, shift, o, cch, tot);
  };
  auto dcrun = [&](int k, const _Float16* in, _Float16* midbuf, _Float16* outbuf) {
    const DCfg& C = dc[k];
    conv(in, wk[k * 2 + 0], raw, C.cin, C.mid, C.Hs);
    bnrelu(P(8 * k + 2), P(8 * k + 3), midbuf, C.mid, C.Hs);
    conv(midbuf, wk[k * 2 + 1], raw, C.mid, C.cout, C.Hs);
    bnrelu(P(8 * k + 6), P(8 * k + 7), outbuf, C.cout, C.Hs);
  };
  auto maxpool = [&](const _Float16* in, _Float16* out, int cch, int Hs) {
    size_t tot = (size_t)Nimg * cch * (Hs / 2) * (Hs / 2);
    k_maxpool<<<grid1(tot), 256, 0, stream>>>(in, out, cch, Hs, Hs, tot);
  };
  auto upcat = [&](const _Float16* x2, const _Float16* x1, _Float16* cat, int C2, int C1, int Ho) {
    size_t tot = (size_t)Nimg * (C1 + C2) * Ho * Ho;
    k_upcat<<<grid1(tot), 256, 0, stream>>>(x2, x1, cat, C2, C1, Ho, Ho, tot);
  };

  // ---- pack weights + convert input (idempotent)
  for (int k = 0; k < 9; ++k) {
    pack(P(8 * k + 0), wk[k * 2 + 0], dc[k].cin, dc[k].mid);
    pack(P(8 * k + 4), wk[k * 2 + 1], dc[k].mid, dc[k].cout);
  }
  pack(P(IDX_GRU_W), wk[18], 1024, 1536);
  {
    size_t tot = (size_t)Nimg * HW1 * 32;
    k_x_to_nhwc<<<grid1(tot), 256, 0, stream>>>(x_f32, xin, 13, 32, (int)HW1, tot);
  }

  // ---- encoder
  dcrun(0, xin, actA, x1s);             // x1: 64 @ 256
  maxpool(x1s, actA, 64, 256);          // 64 @ 128
  dcrun(1, actA, actB, x2s);            // x2: 128 @ 128
  maxpool(x2s, actA, 128, 128);         // 128 @ 64
  dcrun(2, actA, actB, x3s);            // x3: 256 @ 64
  maxpool(x3s, actA, 256, 64);          // 256 @ 32
  dcrun(3, actA, actB, x4s);            // x4: 512 @ 32

  // ---- V2V fusion @ 32x32, C=512
  {
    const int Hs = 32, C = 512;
    size_t tot_rot = (size_t)B * NAGENT * NAGENT * C * Hs * Hs;
    k_warp_rot<<<grid1(tot_rot), 256, 0, stream>>>(x4s, tm, raw, B, C, Hs, Hs, tot_rot);
    size_t tot_f = (size_t)Nimg * C * Hs * Hs;
    k_fuse_cat<<<grid1(tot_f), 256, 0, stream>>>(x4s, raw, tm, actA, B, C, Hs, Hs, tot_f);
    conv(actA, wk[18], raw, 1024, 1536, Hs);
    k_gru_gate<<<grid1(tot_f), 256, 0, stream>>>(raw, P(IDX_GRU_BIH), P(IDX_GRU_BHH),
                                                 fms, B, C, Hs * Hs, tot_f);
  }

  // ---- bottleneck + decoder
  maxpool(fms, actA, 512, 32);          // 512 @ 16
  dcrun(4, actA, actB, x5s);            // x5: 512 @ 16
  upcat(fms, x5s, actA, 512, 512, 32);  // 1024 @ 32
  dcrun(5, actA, actB, actA);           // 256 @ 32
  upcat(x3s, actA, actB, 256, 256, 64); // 512 @ 64
  dcrun(6, actB, actA, actB);           // 128 @ 64
  upcat(x2s, actB, actA, 128, 128, 128);// 256 @ 128
  dcrun(7, actA, actB, actA);           // 64 @ 128
  upcat(x1s, actA, actB, 64, 64, 256);  // 128 @ 256
  dcrun(8, actB, actA, actB);           // 64 @ 256

  // ---- output head
  {
    size_t tot = (size_t)Nimg * HW1;
    k_outc<<<grid1(tot), 256, 0, stream>>>(actB, P(IDX_OUTC_W), P(IDX_OUTC_B),
                                           (float*)d_out, (int)HW1, tot);
  }
}
